// InvLocalPatOrientConvolution_47691316855082
// MI455X (gfx1250) — matrix operand order repaired and checked by reading, not compiled, not run on hardware
//
#include <hip/hip_runtime.h>
#include <hip/hip_bf16.h>

// ---------------------------------------------------------------------------
// Problem constants (from reference)
// ---------------------------------------------------------------------------
#define LBW    6
#define KGRID  32           // grid points per Euler angle
#define BATCH  8
#define CIN    16
#define COUT   32
#define TOTAL  286          // sum (2l+1)^2, l<6
#define TPAD   288          // TOTAL padded to multiple of 32 (WMMA K-step)
#define MROWS  256          // BATCH*COUT
#define NROT   32768        // KGRID^3
#define NTILES (NROT / 16)  // 2048 N-tiles of 16 rotations
#define ICHUNKS (TPAD / 8)  // 36 chunks of 8 halves per wig row

typedef _Float16 v16h __attribute__((ext_vector_type(16)));
typedef _Float16 v8h  __attribute__((ext_vector_type(8)));
typedef float    v8f  __attribute__((ext_vector_type(8)));

// ---------------------------------------------------------------------------
// Kernel 1: y[b,o,i] = sum_c x[b,c,i] * w[o,c,i]   ->  f16 Y[256 x 288]
// ---------------------------------------------------------------------------
__global__ __launch_bounds__(256) void y_kernel(
    const float* __restrict__ x, const float* __restrict__ w,
    _Float16* __restrict__ Y) {
  int e   = blockIdx.x * 256 + threadIdx.x;     // < 256*288
  int i   = e % TPAD;
  int row = e / TPAD;                            // row = b*32 + o
  float s = 0.0f;
  if (i < TOTAL) {
    int b = row >> 5;
    int o = row & 31;
    const float* xp = x + (size_t)(b * CIN) * TOTAL + i;
    const float* wp = w + (size_t)(o * CIN) * TOTAL + i;
#pragma unroll
    for (int c = 0; c < CIN; ++c)
      s += xp[(size_t)c * TOTAL] * wp[(size_t)c * TOTAL];
  }
  Y[e] = (_Float16)s;
}

// ---------------------------------------------------------------------------
// Kernel 2: wig[n,i], n = a*1024 + b*32 + g   ->  f16 WIG[32768 x 288]
//   wig = ca1[mi,a]*d1[b,i]*cg1[ni,g] + ca2[mi,a]*d2[b,i]*cg2[ni,g]
// Each thread produces 8 consecutive i for one rotation n -> one b128 store.
// ---------------------------------------------------------------------------
__global__ __launch_bounds__(256) void wig_kernel(
    const float* __restrict__ d1,  const float* __restrict__ d2,
    const float* __restrict__ ca1, const float* __restrict__ cg1,
    const float* __restrict__ ca2, const float* __restrict__ cg2,
    const int*   __restrict__ mi,  const int*   __restrict__ ni,
    _Float16* __restrict__ WIG) {
  int id = blockIdx.x * 256 + threadIdx.x;       // < 32768*36
  int n     = id / ICHUNKS;
  int ibase = (id - n * ICHUNKS) * 8;

  int p = n >> 10;          // alpha index
  int q = (n >> 5) & 31;    // beta index
  int r = n & 31;           // gamma index

  const float* d1q = d1 + (size_t)q * TOTAL;
  const float* d2q = d2 + (size_t)q * TOTAL;

  v8h vals;
#pragma unroll
  for (int j = 0; j < 8; ++j) {
    int i = ibase + j;
    float v = 0.0f;
    if (i < TOTAL) {
      int m  = mi[i];
      int nn = ni[i];
      v = ca1[m * KGRID + p] * d1q[i] * cg1[nn * KGRID + r] +
          ca2[m * KGRID + p] * d2q[i] * cg2[nn * KGRID + r];
    }
    vals[j] = (_Float16)v;
  }
  *(v8h*)(WIG + (size_t)n * TPAD + ibase) = vals;   // 16B-aligned b128 store
}

// ---------------------------------------------------------------------------
// Kernel 3: resp = Y (256x288) @ WIG^T (288x32768) via v_wmma_f32_16x16x32_f16,
// fused with per-row max over this block's 16 rotations.
// Grid: 2048 blocks (one N-tile each), 256 threads = 8 waves.
// Wave w owns M-tiles {2w, 2w+1}  -> rows 32w..32w+31 (no collisions).
// ---------------------------------------------------------------------------
__global__ __launch_bounds__(256) void gemm_max_kernel(
    const _Float16* __restrict__ Y, const _Float16* __restrict__ WIG,
    float* __restrict__ partial) {
  const int lane = threadIdx.x & 31;
  const int wave = threadIdx.x >> 5;
  const int nt   = blockIdx.x;            // N-tile: rotations nt*16 .. nt*16+15
  const int col  = lane & 15;
  const int hi   = lane >> 4;             // half-wave selector

  // B fragment source: column n of B == row (nt*16+col) of WIG; per-lane data
  // is 16 contiguous K-values starting at hi*16 (32B aligned).
  const _Float16* brow = WIG + (size_t)(nt * 16 + col) * TPAD + hi * 16;

  const int mt0   = wave * 2;             // two M-tiles per wave
  const _Float16* arow0 = Y + (size_t)(mt0 * 16 + col) * TPAD + hi * 16;
  const _Float16* arow1 = arow0 + (size_t)16 * TPAD;

  v8f acc0 = {};
  v8f acc1 = {};

#pragma unroll
  for (int k = 0; k < TPAD; k += 32) {
    v16h bfrag = *(const v16h*)(brow + k);

    // A fragment: halves [0..7] = K k+hi*16.., halves [8..15] = K k+hi*16+8..
    v8h a0lo = *(const v8h*)(arow0 + k);
    v8h a0hi = *(const v8h*)(arow0 + k + 8);
    v16h af0 = __builtin_shufflevector(a0lo, a0hi,
        0, 1, 2, 3, 4, 5, 6, 7, 8, 9, 10, 11, 12, 13, 14, 15);
    acc0 = __builtin_amdgcn_wmma_f32_16x16x32_f16(
        false, af0, false, bfrag, (short)0, acc0, false, false);

    v8h a1lo = *(const v8h*)(arow1 + k);
    v8h a1hi = *(const v8h*)(arow1 + k + 8);
    v16h af1 = __builtin_shufflevector(a1lo, a1hi,
        0, 1, 2, 3, 4, 5, 6, 7, 8, 9, 10, 11, 12, 13, 14, 15);
    acc1 = __builtin_amdgcn_wmma_f32_16x16x32_f16(
        false, af1, false, bfrag, (short)0, acc1, false, false);
  }

  // Row-max over the 16 columns of each accumulator tile.
  // Acc layout: VGPR v, lanes 0-15 -> row v, lanes 16-31 -> row v+8; col = lane&15.
  // partial layout: [row][nt] so the final reduction reads coalesced rows.
#pragma unroll
  for (int v = 0; v < 8; ++v) {
    float m0 = acc0[v];
    m0 = fmaxf(m0, __shfl_xor(m0, 8, 32));
    m0 = fmaxf(m0, __shfl_xor(m0, 4, 32));
    m0 = fmaxf(m0, __shfl_xor(m0, 2, 32));
    m0 = fmaxf(m0, __shfl_xor(m0, 1, 32));
    float m1 = acc1[v];
    m1 = fmaxf(m1, __shfl_xor(m1, 8, 32));
    m1 = fmaxf(m1, __shfl_xor(m1, 4, 32));
    m1 = fmaxf(m1, __shfl_xor(m1, 2, 32));
    m1 = fmaxf(m1, __shfl_xor(m1, 1, 32));
    if (col == 0) {
      // rows are wave-exclusive -> plain stores, no atomics
      partial[(size_t)(mt0 * 16 + v + hi * 8) * NTILES + nt]       = m0;
      partial[(size_t)((mt0 + 1) * 16 + v + hi * 8) * NTILES + nt] = m1;
    }
  }
}

// ---------------------------------------------------------------------------
// Kernel 4: out[row] = max over 2048 N-tiles of partial[row, nt]
// 256 blocks (one per row), coalesced row scan + shuffle/LDS tree reduce.
// ---------------------------------------------------------------------------
__global__ __launch_bounds__(256) void final_max_kernel(
    const float* __restrict__ partial, float* __restrict__ out) {
  const int row = blockIdx.x;
  const int t   = threadIdx.x;
  const float* pr = partial + (size_t)row * NTILES;

  float m = -3.402823466e38f;
#pragma unroll
  for (int j = t; j < NTILES; j += 256)
    m = fmaxf(m, pr[j]);

  // wave reduce (wave32)
  m = fmaxf(m, __shfl_xor(m, 16, 32));
  m = fmaxf(m, __shfl_xor(m, 8, 32));
  m = fmaxf(m, __shfl_xor(m, 4, 32));
  m = fmaxf(m, __shfl_xor(m, 2, 32));
  m = fmaxf(m, __shfl_xor(m, 1, 32));

  __shared__ float sm[8];
  if ((t & 31) == 0) sm[t >> 5] = m;
  __syncthreads();
  if (t < 32) {
    float v = (t < 8) ? sm[t] : -3.402823466e38f;
    v = fmaxf(v, __shfl_xor(v, 4, 32));
    v = fmaxf(v, __shfl_xor(v, 2, 32));
    v = fmaxf(v, __shfl_xor(v, 1, 32));
    if (t == 0) out[row] = v;
  }
}

// ---------------------------------------------------------------------------
// Launch
// ---------------------------------------------------------------------------
extern "C" void kernel_launch(void* const* d_in, const int* in_sizes, int n_in,
                              void* d_out, int out_size, void* d_ws, size_t ws_size,
                              hipStream_t stream) {
  const float* x   = (const float*)d_in[0];
  const float* w   = (const float*)d_in[1];
  const float* d1  = (const float*)d_in[2];
  const float* d2  = (const float*)d_in[3];
  const float* ca1 = (const float*)d_in[4];
  const float* cg1 = (const float*)d_in[5];
  const float* ca2 = (const float*)d_in[6];
  const float* cg2 = (const float*)d_in[7];
  const int*   mi  = (const int*)d_in[8];
  const int*   ni  = (const int*)d_in[9];

  // Workspace layout (all offsets 256B-aligned):
  //   WIG  f16 [32768 x 288]  = 18,874,368 B
  //   Y    f16 [  256 x 288]  =    147,456 B
  //   part f32 [  256 x 2048] =  2,097,152 B
  char* ws = (char*)d_ws;
  _Float16* WIG = (_Float16*)ws;
  _Float16* Yb  = (_Float16*)(ws + (size_t)NROT * TPAD * sizeof(_Float16));
  float* partial = (float*)(ws + (size_t)NROT * TPAD * sizeof(_Float16)
                               + (size_t)MROWS * TPAD * sizeof(_Float16));
  float* out = (float*)d_out;

  y_kernel<<<(MROWS * TPAD) / 256, 256, 0, stream>>>(x, w, Yb);
  wig_kernel<<<(NROT * ICHUNKS) / 256, 256, 0, stream>>>(
      d1, d2, ca1, cg1, ca2, cg2, mi, ni, WIG);
  gemm_max_kernel<<<NTILES, 256, 0, stream>>>(Yb, WIG, partial);
  final_max_kernel<<<MROWS, 256, 0, stream>>>(partial, out);
}
